// Position_attention_11218454577343
// MI455X (gfx1250) — matrix operand order repaired and checked by reading, not compiled, not run on hardware
//
#include <hip/hip_runtime.h>
#include <hip/hip_bf16.h>

// Problem dims (fixed by the reference)
#define BB   4
#define CC   256
#define MIDD 32
#define NN   4096   // 64*64

typedef __attribute__((ext_vector_type(8)))  float  v8f;
typedef __attribute__((ext_vector_type(8)))  __bf16 v8bf;
typedef __attribute__((ext_vector_type(16))) __bf16 v16bf;

// DPP16 lane exchange (VALU, no LDS): ctrl must be an immediate.
//  0xB1 = quad_perm(1,0,3,2)  -> xor 1
//  0x4E = quad_perm(2,3,0,1)  -> xor 2
//  0x141 = row_half_mirror    -> exchange quads within 8 (after quads uniform)
//  0x140 = row_mirror         -> exchange 8-groups within 16
#define DPP_XCHG(v, ctrl) \
    __int_as_float(__builtin_amdgcn_update_dpp(0, __float_as_int(v), (ctrl), 0xF, 0xF, true))

// ---------------------------------------------------------------------------
// Kernel 1: 1x1-conv projections.
//   Q[b][n][mid], K[b][n][mid]  (bf16, 64B rows -> match WMMA A/B lane patterns)
//   V[b][c][n]                  (bf16, c-major -> PV B-tile contiguous per lane)
// ---------------------------------------------------------------------------
__global__ void pa_proj_kernel(const float* __restrict__ x,
                               const float* __restrict__ Wq, const float* __restrict__ bq,
                               const float* __restrict__ Wk, const float* __restrict__ bk,
                               const float* __restrict__ Wv, const float* __restrict__ bv,
                               __bf16* __restrict__ Qb, __bf16* __restrict__ Kb,
                               __bf16* __restrict__ Vb)
{
    const int TN = 32;                       // spatial columns per block
    __shared__ float xs[CC * TN];            // 32 KB x-tile [c][n]

    const int blk = blockIdx.x;              // 0 .. BB*(NN/TN)-1
    const int b   = blk / (NN / TN);
    const int n0  = (blk % (NN / TN)) * TN;
    const int t   = threadIdx.x;             // 0..255

    const float* xb = x + (size_t)b * CC * NN;
    #pragma unroll
    for (int r = 0; r < (CC * TN) / 256; ++r) {
        int f = t + 256 * r;
        int c = f / TN, n = f % TN;
        xs[f] = xb[(size_t)c * NN + n0 + n];
    }
    __syncthreads();

    const int n     = t % TN;
    const int obase = t / TN;                // 0..7

    #pragma unroll
    for (int rr = 0; rr < MIDD / 8; ++rr) {
        int o = obase + 8 * rr;
        float sq = 0.f, sk = 0.f;
        #pragma unroll 4
        for (int c = 0; c < CC; ++c) {
            float xv = xs[c * TN + n];
            sq = fmaf(Wq[o * CC + c], xv, sq);
            sk = fmaf(Wk[o * CC + c], xv, sk);
        }
        sq += bq[o];
        sk += bk[o];
        size_t row = ((size_t)b * NN + n0 + n) * MIDD;
        Qb[row + o] = (__bf16)sq;
        Kb[row + o] = (__bf16)sk;
    }

    #pragma unroll
    for (int rr = 0; rr < CC / 8; ++rr) {
        int o = obase + 8 * rr;
        float sv = 0.f;
        #pragma unroll 4
        for (int c = 0; c < CC; ++c)
            sv = fmaf(Wv[o * CC + c], xs[c * TN + n], sv);
        sv += bv[o];
        Vb[((size_t)b * CC + o) * NN + n0 + n] = (__bf16)sv;
    }
}

// ---------------------------------------------------------------------------
// Kernel 2: flash attention. 4 waves/block, one wave per 16-query tile.
// K/V tiles staged per-block in LDS (double buffered, software pipelined);
// softmax row reductions via DPP16 butterflies (pure VALU).
// ---------------------------------------------------------------------------
__global__ void __launch_bounds__(128)
pa_attn_kernel(const float* __restrict__ x,
               const __bf16* __restrict__ Qb, const __bf16* __restrict__ Kb,
               const __bf16* __restrict__ Vb,
               const float* __restrict__ gamma, float* __restrict__ out)
{
    constexpr int WPB = 4;                        // waves per block
    __shared__ __align__(128) __bf16 Kt[2][32 * 32];        //  2 x 2 KB
    __shared__ __align__(128) __bf16 Vt[2][CC * 32];        //  2 x 16 KB
    __shared__ __align__(128) __bf16 Pl[WPB][16 * 32];      //  4 x 1 KB

    const int t    = threadIdx.x;                 // 0..127
    const int wave = t >> 5;
    const int lane = t & 31;

    const int tile = blockIdx.x * WPB + wave;     // 0 .. BB*NN/16 - 1
    const int b    = tile / (NN / 16);
    const int n0   = (tile % (NN / 16)) * 16;

    const int half = lane >> 4;
    const int l16  = lane & 15;
    const int o    = half * 8;

    const __bf16* kbase = Kb + (size_t)b * NN * MIDD;
    const __bf16* vbase = Vb + (size_t)b * CC * (size_t)NN;

    // ---- Q A-tile (contiguous k order: k = 16*half + s, matches K B-tile)
    const __bf16* qrow = Qb + ((size_t)b * NN + n0 + l16) * MIDD;
    v16bf qa = *(const v16bf*)(qrow + half * 16);

    // ---- prologue: stage K/V tile 0 into LDS buffer 0
    {
        const uint4* vs0 = (const uint4*)(vbase + (size_t)t * NN);
        const uint4* vs1 = (const uint4*)(vbase + (size_t)(t + 128) * NN);
        uint4* vd0 = (uint4*)(&Vt[0][t * 32]);
        uint4* vd1 = (uint4*)(&Vt[0][(t + 128) * 32]);
        #pragma unroll
        for (int r = 0; r < 4; ++r) { vd0[r] = vs0[r]; vd1[r] = vs1[r]; }
        if (t < 32) {
            const uint4* ks = (const uint4*)(kbase + (size_t)t * MIDD);
            uint4* kd = (uint4*)(&Kt[0][t * 32]);
            #pragma unroll
            for (int r = 0; r < 4; ++r) kd[r] = ks[r];
        }
    }
    __syncthreads();

    v8f acc[16];
    #pragma unroll
    for (int tt = 0; tt < 16; ++tt) { v8f z = {}; acc[tt] = z; }
    float mrow[8], lrow[8];
    #pragma unroll
    for (int g = 0; g < 8; ++g) { mrow[g] = -1e30f; lrow[g] = 0.f; }

    const v8f cz = {};

    for (int it = 0; it < NN / 32; ++it) {
        const int jt  = it * 32;
        const int cur = it & 1, nxt = cur ^ 1;

        // ---- pipelined load-phase for next K/V tile (global -> VGPR)
        uint4 vreg[8], kreg[4];
        const bool have = (it + 1 < NN / 32);
        if (have) {
            const int jn = jt + 32;
            const uint4* vs0 = (const uint4*)(vbase + (size_t)t * NN + jn);
            const uint4* vs1 = (const uint4*)(vbase + (size_t)(t + 128) * NN + jn);
            #pragma unroll
            for (int r = 0; r < 4; ++r) { vreg[r] = vs0[r]; vreg[4 + r] = vs1[r]; }
            if (t < 32) {
                const uint4* ks = (const uint4*)(kbase + (size_t)(jn + t) * MIDD);
                #pragma unroll
                for (int r = 0; r < 4; ++r) kreg[r] = ks[r];
            }
            if (jn + 32 < NN)   // j+2 tile -> global_prefetch_b8
                __builtin_prefetch(vbase + (size_t)t * NN + jn + 32, 0, 0);
        }

        // ---- score WMMAs from LDS K tile
        const __bf16* kt = Kt[cur];
        v16bf kb0 = *(const v16bf*)(kt + l16 * 32 + half * 16);
        v16bf kb1 = *(const v16bf*)(kt + (16 + l16) * 32 + half * 16);
        v8f s0 = __builtin_amdgcn_wmma_f32_16x16x32_bf16(false, qa, false, kb0,
                                                         (short)0, cz, false, false);
        v8f s1 = __builtin_amdgcn_wmma_f32_16x16x32_bf16(false, qa, false, kb1,
                                                         (short)0, cz, false, false);

        // ---- online softmax; row reductions = DPP16 butterfly (VALU only)
        float scale[8];
        __bf16* plds = Pl[wave];
        #pragma unroll
        for (int g = 0; g < 8; ++g) {
            float a0 = s0[g], a1 = s1[g];
            float r = fmaxf(a0, a1);
            r = fmaxf(r, DPP_XCHG(r, 0xB1));
            r = fmaxf(r, DPP_XCHG(r, 0x4E));
            r = fmaxf(r, DPP_XCHG(r, 0x141));
            r = fmaxf(r, DPP_XCHG(r, 0x140));     // row max over 16 lanes
            float mnew = fmaxf(mrow[g], r);
            float sc   = __expf(mrow[g] - mnew);
            float p0   = __expf(a0 - mnew);
            float p1   = __expf(a1 - mnew);
            float ls   = p0 + p1;
            ls += DPP_XCHG(ls, 0xB1);
            ls += DPP_XCHG(ls, 0x4E);
            ls += DPP_XCHG(ls, 0x141);
            ls += DPP_XCHG(ls, 0x140);            // row sum over 16 lanes
            lrow[g]  = lrow[g] * sc + ls;
            mrow[g]  = mnew;
            scale[g] = sc;
            int i = g + half * 8;                 // P -> LDS, [i][j] row-major
            plds[i * 32 +      l16] = (__bf16)p0;
            plds[i * 32 + 16 + l16] = (__bf16)p1;
        }
        asm volatile("s_wait_dscnt 0" ::: "memory");

        // ---- P readback in A-matrix layout (contiguous k order, matches V B-tile)
        const __bf16* prow = plds + l16 * 32;
        v16bf pa = *(const v16bf*)(prow + half * 16);

        // ---- pipelined store-phase: flush staged regs to next LDS buffer
        if (have) {
            uint4* vd0 = (uint4*)(&Vt[nxt][t * 32]);
            uint4* vd1 = (uint4*)(&Vt[nxt][(t + 128) * 32]);
            #pragma unroll
            for (int r = 0; r < 4; ++r) { vd0[r] = vreg[r]; vd1[r] = vreg[4 + r]; }
            if (t < 32) {
                uint4* kd = (uint4*)(&Kt[nxt][t * 32]);
                #pragma unroll
                for (int r = 0; r < 4; ++r) kd[r] = kreg[r];
            }
        }

        // ---- 16 c-tiles: rescale accumulators, WMMA-accumulate P x V (LDS)
        const __bf16* vt = Vt[cur];
        #pragma unroll
        for (int tt = 0; tt < 16; ++tt) {
            v16bf vb = *(const v16bf*)(vt + (tt * 16 + l16) * 32 + half * 16);
            v8f a = acc[tt];
            #pragma unroll
            for (int g = 0; g < 8; ++g) a[g] *= scale[g];
            acc[tt] = __builtin_amdgcn_wmma_f32_16x16x32_bf16(false, pa, false, vb,
                                                              (short)0, a, false, false);
        }

        __syncthreads();
    }

    // ---- epilogue: out = gamma * (O / l) + x   (32B contiguous per lane)
    const float gm = gamma[0];
    float rl[8];
    #pragma unroll
    for (int g = 0; g < 8; ++g) rl[g] = 1.0f / lrow[g];

    #pragma unroll
    for (int tt = 0; tt < 16; ++tt) {
        int c = tt * 16 + l16;                    // D layout: lane -> channel
        size_t base = ((size_t)b * CC + c) * NN + n0 + half * 8;
        float4 x0 = *(const float4*)(x + base);
        float4 x1 = *(const float4*)(x + base + 4);
        float4 o0, o1;
        o0.x = fmaf(gm, acc[tt][0] * rl[0], x0.x);
        o0.y = fmaf(gm, acc[tt][1] * rl[1], x0.y);
        o0.z = fmaf(gm, acc[tt][2] * rl[2], x0.z);
        o0.w = fmaf(gm, acc[tt][3] * rl[3], x0.w);
        o1.x = fmaf(gm, acc[tt][4] * rl[4], x1.x);
        o1.y = fmaf(gm, acc[tt][5] * rl[5], x1.y);
        o1.z = fmaf(gm, acc[tt][6] * rl[6], x1.z);
        o1.w = fmaf(gm, acc[tt][7] * rl[7], x1.w);
        *(float4*)(out + base)     = o0;
        *(float4*)(out + base + 4) = o1;
    }
}

// ---------------------------------------------------------------------------
extern "C" void kernel_launch(void* const* d_in, const int* in_sizes, int n_in,
                              void* d_out, int out_size, void* d_ws, size_t ws_size,
                              hipStream_t stream)
{
    const float* x     = (const float*)d_in[0];
    const float* Wq    = (const float*)d_in[1];
    const float* bq    = (const float*)d_in[2];
    const float* Wk    = (const float*)d_in[3];
    const float* bk    = (const float*)d_in[4];
    const float* Wv    = (const float*)d_in[5];
    const float* bv    = (const float*)d_in[6];
    const float* gamma = (const float*)d_in[7];
    float* out = (float*)d_out;

    __bf16* Qb = (__bf16*)d_ws;
    __bf16* Kb = Qb + (size_t)BB * NN * MIDD;
    __bf16* Vb = Kb + (size_t)BB * NN * MIDD;

    pa_proj_kernel<<<BB * (NN / 32), 256, 0, stream>>>(x, Wq, bq, Wk, bk, Wv, bv,
                                                       Qb, Kb, Vb);
    pa_attn_kernel<<<(BB * NN / 16) / 4, 128, 0, stream>>>(x, Qb, Kb, Vb, gamma, out);
}
